// FocusedLinearAttn_1778116460958
// MI455X (gfx1250) — matrix operand order repaired and checked by reading, not compiled, not run on hardware
//
#include <hip/hip_runtime.h>

// ---------------------------------------------------------------------------
// FocusedLinearAttention for MI455X (gfx1250, wave32, WMMA + async-to-LDS)
// Heavy matmuls: v_wmma_f32_16x16x32_bf16, tiles staged with
// global_load_async_to_lds_b128 (ASYNCcnt) and double-buffered LDS.
// Steady-state loop is branch-free (last K-step peeled).
// ---------------------------------------------------------------------------

typedef __attribute__((ext_vector_type(16))) __bf16 v16bf;
typedef __attribute__((ext_vector_type(8)))  float  v8f;

struct alignas(16) U4 { unsigned int a, b, c, d; };
union FragBF { v16bf v; U4 q[2]; };

static constexpr int Nn    = 8;
static constexpr int Cc    = 512;
static constexpr int HEADS = 8;
static constexpr int HD    = 64;          // head dim
static constexpr int Ls    = 64 * 64;     // 4096 (H*W)
static constexpr int Mrows = Nn * Ls;     // 32768
static constexpr int TC    = 3 * Cc;      // 1536
static constexpr int BH    = Nn * HEADS;  // 64

__device__ __forceinline__ unsigned short f2bf(float f) {
  unsigned int u = __float_as_uint(f);
  unsigned int r = u + 0x7fffu + ((u >> 16) & 1u);   // round-to-nearest-even
  return (unsigned short)(r >> 16);
}

// Async 16B global->LDS copy (gfx1250, tracked by ASYNCcnt).
// Low 32 bits of a flat shared-aperture address are the LDS byte offset.
__device__ __forceinline__ void async_b128(void* lds, const void* gptr) {
  const unsigned l = (unsigned)(unsigned long long)lds;
  asm volatile("global_load_async_to_lds_b128 %0, %1, off"
               :: "v"(l), "v"(gptr) : "memory");
}

template <int N>
__device__ __forceinline__ void wait_async() {
  if (N == 0)      asm volatile("s_wait_asynccnt 0x0" ::: "memory");
  else if (N == 3) asm volatile("s_wait_asynccnt 0x3" ::: "memory");
  else if (N == 4) asm volatile("s_wait_asynccnt 0x4" ::: "memory");
  else             asm volatile("s_wait_asynccnt 0x0" ::: "memory");
}

// ---------------------------------------------------------------------------
// f32 -> bf16 bulk convert
// ---------------------------------------------------------------------------
__global__ void cvt_f32_bf16(const float* __restrict__ s,
                             unsigned short* __restrict__ d, size_t n) {
  size_t i = (size_t)blockIdx.x * blockDim.x + threadIdx.x;
  const size_t st = (size_t)gridDim.x * blockDim.x;
  for (; i < n; i += st) d[i] = f2bf(s[i]);
}

// ---------------------------------------------------------------------------
// f32 [rows][cols] -> bf16 transposed [cols][rows]  (64x64 LDS tiles)
// ---------------------------------------------------------------------------
__global__ __launch_bounds__(256)
void cvt_transpose_f32_bf16(const float* __restrict__ src,
                            unsigned short* __restrict__ dst,
                            int rows, int cols) {
  __shared__ float tile[64][65];
  const int c0 = blockIdx.x * 64, r0 = blockIdx.y * 64;
#pragma unroll
  for (int i = 0; i < 16; ++i) {
    const int idx = threadIdx.x + i * 256;
    const int r = idx >> 6, c = idx & 63;
    tile[r][c] = src[(size_t)(r0 + r) * cols + c0 + c];
  }
  __syncthreads();
#pragma unroll
  for (int i = 0; i < 16; ++i) {
    const int idx = threadIdx.x + i * 256;
    const int c = idx >> 6, r = idx & 63;
    dst[(size_t)(c0 + c) * rows + r0 + r] = f2bf(tile[r][c]);
  }
}

// ---------------------------------------------------------------------------
// Generic tiled bf16 WMMA GEMM:  C = A(MxK) * B(KxN), B given pre-transposed
// as BT(NxK). Double-buffered LDS, tiles staged with async global->LDS b128.
// Wave tile = 32 x (BN/WC): 2 x TN grid of v_wmma_f32_16x16x32_bf16.
// EPI: 1 = f32 store + bias, 2 = residual gate, 3 = bf16 store,
//      4 = bf16 store transposed (C^T).
// Batched over blockIdx.z; C batch offset = (z/zdiv)*cs1 + (z%zdiv)*cs2.
// ---------------------------------------------------------------------------
template<int BM, int BN, int WR, int WC, int EPI>
__global__ __launch_bounds__(WR * WC * 32)
void gemm_bf16(const unsigned short* __restrict__ A, long long sA,
               const unsigned short* __restrict__ BT, long long sB,
               void* __restrict__ Cv, int zdiv, long long cs1, long long cs2,
               int K, int lda, int ldbt, int ldc,
               const float* __restrict__ bias, const float* __restrict__ gamma,
               const float* __restrict__ beta, const float* __restrict__ resid,
               int ldr) {
  constexpr int NT   = WR * WC * 32;
  constexpr int TN   = BN / (WC * 16);       // WMMA tiles per wave in N
  constexpr int SEGS = (BM + BN) * 4;        // 16B chunks per 32-deep tile pair
  constexpr int PER  = SEGS / NT;            // async ops per thread per stage
  static_assert((BM * 4) % NT == 0, "uniform A/B chunk split");
  static_assert(BM == WR * 32, "wave rows cover BM");

  __shared__ unsigned short As[2][BM][40];   // 40 = 32 + 8 pad (16B aligned)
  __shared__ unsigned short Bs[2][BN][40];

  const int tid  = threadIdx.x;
  const int lane = tid & 31;
  const int wave = tid >> 5;
  const int wr   = wave / WC;
  const int wc   = wave % WC;
  const int bm0  = blockIdx.x * BM;
  const int bn0  = blockIdx.y * BN;

  const unsigned short* Ab = A  + (size_t)blockIdx.z * (size_t)sA;
  const unsigned short* Bb = BT + (size_t)blockIdx.z * (size_t)sB;
  const size_t coff = (size_t)((long long)(blockIdx.z / zdiv) * cs1 +
                               (long long)(blockIdx.z % zdiv) * cs2);

  v8f acc[2][TN] = {};
  const int kb   = (lane & 16) ? 8 : 0;    // A fragment K base
  const int kb2  = (lane & 16) ? 16 : 0;   // unused (B fragment also uses kb)
  const int mrow = lane & 15;

  // issue one tile-pair's async copies into LDS buffer `b`
  auto stage = [&](int k0, int b) {
#pragma unroll
    for (int i = 0; i < PER; ++i) {
      const int base = i * NT;
      if (base < BM * 4) {                  // A chunk (uniform per i)
        const int s = base + tid;
        const int r = s >> 2, c = (s & 3) * 8;
        async_b128(&As[b][r][c], Ab + (size_t)(bm0 + r) * lda + k0 + c);
      } else {                              // B chunk
        const int s = base - BM * 4 + tid;
        const int r = s >> 2, c = (s & 3) * 8;
        async_b128(&Bs[b][r][c], Bb + (size_t)(bn0 + r) * ldbt + k0 + c);
      }
    }
  };

  auto compute = [&](int b) {
    FragBF aF[2], bF[TN];
#pragma unroll
    for (int tt = 0; tt < 2; ++tt) {
      const int m = wr * 32 + tt * 16 + mrow;
      aF[tt].q[0] = *(const U4*)&As[b][m][kb];
      aF[tt].q[1] = *(const U4*)&As[b][m][16 + kb];
    }
#pragma unroll
    for (int tn = 0; tn < TN; ++tn) {
      const int nloc = wc * (TN * 16) + tn * 16 + mrow;
      bF[tn].q[0] = *(const U4*)&Bs[b][nloc][kb];
      bF[tn].q[1] = *(const U4*)&Bs[b][nloc][16 + kb];
    }
#pragma unroll
    for (int tm = 0; tm < 2; ++tm)
#pragma unroll
      for (int tn = 0; tn < TN; ++tn)
        acc[tm][tn] = __builtin_amdgcn_wmma_f32_16x16x32_bf16(
            false, aF[tm].v, false, bF[tn].v, (short)0, acc[tm][tn],
            false, false);
  };

  // software pipeline: prefetch next tile while computing current
  stage(0, 0);
  int buf = 0;
  int k0 = 0;
  for (; k0 + 32 < K; k0 += 32) {           // branch-free steady state
    stage(k0 + 32, buf ^ 1);
    wait_async<PER>();                      // previous batch landed (in-order)
    __syncthreads();
    compute(buf);
    __syncthreads();
    buf ^= 1;
  }
  wait_async<0>();                          // last tile
  __syncthreads();
  compute(buf);

  // epilogue: C/D layout = VGPR r, lanes 0-15 -> M=r, lanes 16-31 -> M=8+r
#pragma unroll
  for (int tm = 0; tm < 2; ++tm) {
    const int mb = bm0 + wr * 32 + tm * 16 + ((lane & 16) ? 8 : 0);
#pragma unroll
    for (int tn = 0; tn < TN; ++tn) {
      const int nn = bn0 + wc * (TN * 16) + tn * 16 + mrow;
#pragma unroll
      for (int r = 0; r < 8; ++r) {
        const int m = mb + r;
        const float v = acc[tm][tn][r];
        if (EPI == 1) {
          ((float*)Cv)[coff + (size_t)m * ldc + nn] = v + bias[nn];
        } else if (EPI == 2) {
          const float g  = gamma[nn];
          const float sg = 1.0f / (1.0f + __expf(-g));
          ((float*)Cv)[coff + (size_t)m * ldc + nn] =
              resid[(size_t)m * ldr + nn] + (v + bias[nn]) * sg + beta[nn];
        } else if (EPI == 3) {
          ((unsigned short*)Cv)[coff + (size_t)m * ldc + nn] = f2bf(v);
        } else {  // EPI == 4: transposed bf16 store
          ((unsigned short*)Cv)[coff + (size_t)nn * ldc + m] = f2bf(v);
        }
      }
    }
  }
}

// ---------------------------------------------------------------------------
// k softmax stats over sequence dim: per (n,head,col j): max, sum(exp(.-max))
// ---------------------------------------------------------------------------
__global__ __launch_bounds__(256)
void kstats_kernel(const float* __restrict__ qkv, float* __restrict__ kmax,
                   float* __restrict__ ksum) {
  const int bh = blockIdx.x, n = bh >> 3, head = bh & 7;
  const float* kb = qkv + (size_t)n * Ls * TC + Cc + head * HD;
  const int j = threadIdx.x & 63;
  const int rg = threadIdx.x >> 6;  // 0..3
  __shared__ float red[4][64];

  float mloc = -3.4e38f;
  for (int l = rg; l < Ls; l += 4) mloc = fmaxf(mloc, kb[(size_t)l * TC + j]);
  red[rg][j] = mloc;
  __syncthreads();
  const float Mx = fmaxf(fmaxf(red[0][j], red[1][j]), fmaxf(red[2][j], red[3][j]));
  __syncthreads();

  float sloc = 0.f;
  for (int l = rg; l < Ls; l += 4) sloc += __expf(kb[(size_t)l * TC + j] - Mx);
  red[rg][j] = sloc;
  __syncthreads();
  if (rg == 0) {
    const float S = ((red[0][j] + red[1][j]) + (red[2][j] + red[3][j]));
    kmax[bh * 64 + j] = Mx;
    ksum[bh * 64 + j] = S;
  }
}

// ---------------------------------------------------------------------------
// Normalized k, transposed: knT[bh][j][l] = exp(k-max_j)/sum_j  (bf16)
// ---------------------------------------------------------------------------
__global__ __launch_bounds__(256)
void knorm_transpose(const float* __restrict__ qkv,
                     const float* __restrict__ kmax,
                     const float* __restrict__ ksum,
                     unsigned short* __restrict__ knT) {
  const int bh = blockIdx.y, n = bh >> 3, head = bh & 7;
  const int l0 = blockIdx.x * 64;
  __shared__ float tile[64][65];
  __shared__ float cmax[64], cinv[64];
  if (threadIdx.x < 64) {
    cmax[threadIdx.x] = kmax[bh * 64 + threadIdx.x];
    cinv[threadIdx.x] = 1.0f / ksum[bh * 64 + threadIdx.x];
  }
  __syncthreads();
  const float* kb = qkv + (size_t)(n * Ls + l0) * TC + Cc + head * HD;
#pragma unroll
  for (int i = 0; i < 16; ++i) {
    const int idx = threadIdx.x + i * 256;
    const int l = idx >> 6, j = idx & 63;
    tile[l][j] = __expf(kb[(size_t)l * TC + j] - cmax[j]) * cinv[j];
  }
  __syncthreads();
  unsigned short* ob = knT + (size_t)bh * HD * Ls + l0;
#pragma unroll
  for (int i = 0; i < 16; ++i) {
    const int idx = threadIdx.x + i * 256;
    const int j = idx >> 6, l = idx & 63;
    ob[(size_t)j * Ls + l] = f2bf(tile[l][j]);
  }
}

// ---------------------------------------------------------------------------
// v slice, transposed to vT[bh][d][l] bf16 (so GEMM-B is K-contiguous)
// ---------------------------------------------------------------------------
__global__ __launch_bounds__(256)
void v_transpose(const float* __restrict__ qkv, unsigned short* __restrict__ vT) {
  const int bh = blockIdx.y, n = bh >> 3, head = bh & 7;
  const int l0 = blockIdx.x * 64;
  __shared__ float tile[64][65];
  const float* vs = qkv + (size_t)(n * Ls + l0) * TC + 2 * Cc + head * HD;
#pragma unroll
  for (int i = 0; i < 16; ++i) {
    const int idx = threadIdx.x + i * 256;
    const int l = idx >> 6, d = idx & 63;
    tile[l][d] = vs[(size_t)l * TC + d];
  }
  __syncthreads();
  unsigned short* ob = vT + (size_t)bh * HD * Ls + l0;
#pragma unroll
  for (int i = 0; i < 16; ++i) {
    const int idx = threadIdx.x + i * 256;
    const int d = idx >> 6, l = idx & 63;
    ob[(size_t)d * Ls + l] = f2bf(tile[l][d]);
  }
}

// ---------------------------------------------------------------------------
// q row-softmax (over hd=64) -> bf16 [bh][L][64]
// ---------------------------------------------------------------------------
__global__ __launch_bounds__(256)
void q_prep(const float* __restrict__ qkv, unsigned short* __restrict__ qnb) {
  const int bh = blockIdx.x, n = bh >> 3, head = bh & 7;
  const int l = blockIdx.y * 256 + threadIdx.x;
  const float* qr = qkv + (size_t)(n * Ls + l) * TC + head * HD;
  float mx = -3.4e38f;
  for (int j = 0; j < HD; ++j) mx = fmaxf(mx, qr[j]);
  float s = 0.f;
  for (int j = 0; j < HD; ++j) s += __expf(qr[j] - mx);
  const float inv = 1.0f / s;
  unsigned short* qo = qnb + ((size_t)bh * Ls + l) * HD;
  for (int j = 0; j < HD; ++j) qo[j] = f2bf(__expf(qr[j] - mx) * inv);
}

// ---------------------------------------------------------------------------
extern "C" void kernel_launch(void* const* d_in, const int* in_sizes, int n_in,
                              void* d_out, int out_size, void* d_ws,
                              size_t ws_size, hipStream_t stream) {
  const float* x      = (const float*)d_in[0];
  const float* w_qkv  = (const float*)d_in[1];
  const float* b_qkv  = (const float*)d_in[2];
  const float* w_out  = (const float*)d_in[3];
  const float* b_out  = (const float*)d_in[4];
  const float* gamma  = (const float*)d_in[5];
  const float* beta   = (const float*)d_in[6];
  float* out = (float*)d_out;

  char* ws = (char*)d_ws;
  size_t off = 0;
  auto alloc = [&](size_t bytes) {
    size_t o = off;
    off = (off + bytes + 255) & ~(size_t)255;
    return o;
  };
  unsigned short* xb    = (unsigned short*)(ws + alloc((size_t)Mrows * Cc * 2));
  unsigned short* wqbT  = (unsigned short*)(ws + alloc((size_t)Cc * TC * 2));
  unsigned short* wobT  = (unsigned short*)(ws + alloc((size_t)Cc * Cc * 2));
  float*          qkvf  = (float*)(ws + alloc((size_t)Mrows * TC * 4));
  float*          kmax  = (float*)(ws + alloc((size_t)BH * HD * 4));
  float*          ksum  = (float*)(ws + alloc((size_t)BH * HD * 4));
  unsigned short* knT   = (unsigned short*)(ws + alloc((size_t)BH * HD * Ls * 2));
  unsigned short* qnb   = (unsigned short*)(ws + alloc((size_t)BH * Ls * HD * 2));
  unsigned short* vT    = (unsigned short*)(ws + alloc((size_t)BH * HD * Ls * 2));
  unsigned short* ctxT  = (unsigned short*)(ws + alloc((size_t)BH * HD * HD * 2));
  unsigned short* attnb = (unsigned short*)(ws + alloc((size_t)Mrows * Cc * 2));
  (void)ws_size; (void)in_sizes; (void)n_in; (void)out_size;

  // 1) bf16 conversions; weights transposed so GEMM-B is K-contiguous
  cvt_f32_bf16<<<2048, 256, 0, stream>>>(x, xb, (size_t)Mrows * Cc);
  cvt_transpose_f32_bf16<<<dim3(TC / 64, Cc / 64), 256, 0, stream>>>(
      w_qkv, wqbT, Cc, TC);
  cvt_transpose_f32_bf16<<<dim3(Cc / 64, Cc / 64), 256, 0, stream>>>(
      w_out, wobT, Cc, Cc);

  // 2) QKV projection: [32768x512] * [512x1536] + b_qkv -> f32 (big tiles)
  gemm_bf16<128, 128, 4, 2, 1><<<dim3(Mrows / 128, TC / 128, 1), 256, 0,
                                 stream>>>(
      xb, 0, wqbT, 0, qkvf, 1, 0, 0, Cc, Cc, Cc, TC, b_qkv, nullptr, nullptr,
      nullptr, 0);

  // 3) k softmax stats (over L)
  kstats_kernel<<<BH, 256, 0, stream>>>(qkvf, kmax, ksum);

  // 4) normalized k, transposed -> knT[bh][64][L] bf16
  knorm_transpose<<<dim3(Ls / 64, BH), 256, 0, stream>>>(qkvf, kmax, ksum, knT);

  // 5) q row-softmax -> qnb; v transposed -> vT
  q_prep<<<dim3(BH, Ls / 256), 256, 0, stream>>>(qkvf, qnb);
  v_transpose<<<dim3(Ls / 64, BH), 256, 0, stream>>>(qkvf, vT);

  // 6) context^T[bh] = (knT(64xL) * v(Lx64))^T -> bf16 [bh][d][j]
  gemm_bf16<64, 64, 2, 2, 4><<<dim3(1, 1, BH), 128, 0, stream>>>(
      knT, (long long)HD * Ls, vT, (long long)HD * Ls, ctxT, 1,
      (long long)HD * HD, 0, Ls, Ls, Ls, HD, nullptr, nullptr, nullptr,
      nullptr, 0);

  // 7) attn[bh] = qn(Lx64) * ctx(64x64) -> bf16 into [n][l][head*64+j]
  gemm_bf16<128, 64, 4, 2, 3><<<dim3(Ls / 128, 1, BH), 256, 0, stream>>>(
      qnb, (long long)Ls * HD, ctxT, (long long)HD * HD, attnb, HEADS,
      (long long)Ls * Cc, (long long)HD, HD, HD, HD, Cc, nullptr, nullptr,
      nullptr, nullptr, 0);

  // 8) out = x + (attn*w_out + b_out)*sigmoid(gamma) + beta (big tiles, fused)
  gemm_bf16<128, 128, 4, 2, 2><<<dim3(Mrows / 128, Cc / 128, 1), 256, 0,
                                 stream>>>(
      attnb, 0, wobT, 0, out, 1, 0, 0, Cc, Cc, Cc, Cc, b_out, gamma, beta, x,
      Cc);
}
